// BaseBEVBackboneSbnet_75780402971284
// MI455X (gfx1250) — matrix operand order repaired and checked by reading, not compiled, not run on hardware
//
#include <hip/hip_runtime.h>
#include <hip/hip_bf16.h>

typedef __bf16 bf16;
typedef __attribute__((ext_vector_type(16))) __bf16 v16bf;
typedef __attribute__((ext_vector_type(8)))  __bf16 v8bf;
typedef __attribute__((ext_vector_type(8)))  float  v8f;
typedef __attribute__((ext_vector_type(4)))  float  v4f;
typedef __attribute__((ext_vector_type(4)))  int    v4i;

#define WMMA_BF16(a, b, c) \
  __builtin_amdgcn_wmma_f32_16x16x32_bf16(false, (a), false, (b), (short)0, (c), false, false)

#if __has_builtin(__builtin_amdgcn_global_load_async_to_lds_b128) && \
    __has_builtin(__builtin_amdgcn_s_wait_asynccnt)
#define ASYNC_LDS 1
__device__ __forceinline__ void async_b128(const void* g, void* l) {
  __builtin_amdgcn_global_load_async_to_lds_b128(
      (__attribute__((address_space(1))) v4i*)(void*)g,
      (__attribute__((address_space(3))) v4i*)l, 0, 0);
}
#else
#define ASYNC_LDS 0
#endif

// ---------------------------------------------------------------------------
// Mask: 512 tile slots (b*256 + r*16 + c). Flat id == chosen_tile_coords value.
// ---------------------------------------------------------------------------
__global__ __launch_bounds__(32) void k_build_mask(const int* __restrict__ coords,
                                                   int n, int* __restrict__ mask) {
  int t = blockIdx.x * 32 + threadIdx.x;
  if (t >= 512) return;
  int a = 0;
  for (int i = 0; i < n; ++i) a |= (coords[i] == t) ? 1 : 0;
  mask[t] = a;
}

// ---------------------------------------------------------------------------
// Input convert: f32 NCHW [2,64,512,512] -> bf16 HWC [2,512,512,64]
// ---------------------------------------------------------------------------
__global__ __launch_bounds__(256) void k_convert_in(const float* __restrict__ in,
                                                    bf16* __restrict__ out) {
  long tid = (long)blockIdx.x * 256 + threadIdx.x;   // 2*512*512*8 total
  const long NPIX = 512L * 512L;
  if (tid >= 2 * NPIX * 8) return;
  int  cg  = (int)(tid & 7);
  long p   = tid >> 3;                // b*NPIX + pix
  long b   = p >> 18;                 // / NPIX
  long pix = p & (NPIX - 1);
  v8bf o;
  #pragma unroll
  for (int j = 0; j < 8; ++j)
    o[j] = (bf16)in[(b * 64 + cg * 8 + j) * NPIX + pix];
  *(v8bf*)&out[p * 64 + cg * 8] = o;
}

// ---------------------------------------------------------------------------
// Weight pack: fold BN scale into weights, emit bf16 A-tiles in WMMA lane order.
// A-tile (16x32): lane = m + (k>>4)*16, slot = k&15.
// ---------------------------------------------------------------------------
__global__ __launch_bounds__(256) void k_pack(const float* __restrict__ w,
                                              const float* __restrict__ g,
                                              const float* __restrict__ bb,
                                              const float* __restrict__ bm,
                                              const float* __restrict__ bv,
                                              bf16* __restrict__ wpack,
                                              float* __restrict__ bias,
                                              int Cout, int Cin, int ks) {
  int t = blockIdx.x * 256 + threadIdx.x;
  int total = Cout * Cin * ks * ks;
  if (t >= total) return;
  int kw = t % ks;
  int kh = (t / ks) % ks;
  int ci = (t / (ks * ks)) % Cin;
  int co = t / (ks * ks * Cin);
  float inv = g[co] * rsqrtf(bv[co] + 1e-3f);
  if (ci == 0 && kh == 0 && kw == 0) bias[co] = bb[co] - bm[co] * inv;
  float val = w[t] * inv;
  int tap = kh * ks + kw;
  int kc = ci >> 5, kk = ci & 31;
  int mt = co >> 4, mm = co & 15;
  int kchunks = Cin >> 5, mtiles = Cout >> 4;
  long tile = (long)(tap * kchunks + kc) * mtiles + mt;
  int lane = mm + (kk >> 4) * 16;
  int slot = kk & 15;
  wpack[tile * 512 + lane * 16 + slot] = (bf16)val;
}

// ---------------------------------------------------------------------------
// Fused Conv3x3 + bias + ReLU + tile mask. Fully templated.
// One block = one mask tile, WAVES waves covering all 16-channel M-tiles.
// Input halo patch staged in LDS (except L0): async-to-LDS for interior tiles.
// NB=4 N-subtiles share each A-tile global load.
// ---------------------------------------------------------------------------
template<int CIN, int COUT, int INRES, int OUTRES, int STRIDE, bool USE_LDS>
__global__ __launch_bounds__(256)
void k_conv(const bf16* __restrict__ act, const bf16* __restrict__ wpack,
            const float* __restrict__ bias, const int* __restrict__ mask,
            bf16* __restrict__ out) {
  constexpr int TP   = OUTRES / 16;                 // tile side (out px)
  constexpr int TPL  = (TP == 16) ? 4 : (TP == 8) ? 3 : 2;
  constexpr int NSUB = TP * TP / 16;
  constexpr int SPAN = (TP - 1) * STRIDE + 3;       // input patch side
  constexpr int MT   = COUT / 16;
  constexpr int WAVES = (MT < 8) ? MT : 8;
  constexpr int MPW  = MT / WAVES;
  constexpr int KCH  = CIN / 32;
  constexpr int NB   = (NSUB < 4) ? NSUB : 4;
  constexpr int NPASS = NSUB / NB;
  constexpr int NTH  = WAVES * 32;

  __shared__ alignas(32) bf16 patch[USE_LDS ? SPAN * SPAN * CIN : 32];

  const int lane = threadIdx.x;
  const int wave = threadIdx.y;
  const int tid  = wave * 32 + lane;
  const int tileId = blockIdx.x;                    // b*256 + tr*16 + tc
  const int b  = tileId >> 8;
  const int tr = (tileId >> 4) & 15;
  const int tc = tileId & 15;

  if (!mask[tileId]) {                              // inactive: zero whole tile
    v8bf z = {};
    constexpr int NCH = TP * TP * COUT / 8;
    for (int i = tid; i < NCH; i += NTH) {
      int cg  = i % (COUT / 8);
      int pix = i / (COUT / 8);
      int oy = tr * TP + (pix >> TPL);
      int ox = tc * TP + (pix & (TP - 1));
      *(v8bf*)&out[(((long)b * OUTRES + oy) * OUTRES + ox) * COUT + cg * 8] = z;
    }
    return;
  }

  if (wave == 0) {                                  // prefetch weight stream
    for (int t = lane; t < 9; t += 32)
      __builtin_prefetch(&wpack[(long)t * KCH * MT * 512], 0, 0);
  }

  if constexpr (USE_LDS) {                          // stage halo patch, zero-pad
    constexpr int ROWCH = SPAN * CIN / 8;
    constexpr int NCHK  = SPAN * ROWCH;
    const int iy0 = tr * TP * STRIDE - 1;
    const int ix0 = tc * TP * STRIDE - 1;
#if ASYNC_LDS
    const bool interior = (iy0 >= 0) && (ix0 >= 0) &&
                          (iy0 + SPAN <= INRES) && (ix0 + SPAN <= INRES);
    if (interior) {
      for (int c = tid; c < NCHK; c += NTH) {
        int py = c / ROWCH;
        int r  = c - py * ROWCH;
        int px = r / (CIN / 8);
        int cg = r - px * (CIN / 8);
        int iy = iy0 + py, ix = ix0 + px;
        async_b128(&act[(((long)b * INRES + iy) * INRES + ix) * CIN + cg * 8],
                   &patch[(py * SPAN + px) * CIN + cg * 8]);
      }
      __builtin_amdgcn_s_wait_asynccnt(0);
    } else
#endif
    {
      for (int c = tid; c < NCHK; c += NTH) {
        int py = c / ROWCH;
        int r  = c - py * ROWCH;
        int px = r / (CIN / 8);
        int cg = r - px * (CIN / 8);
        int iy = iy0 + py, ix = ix0 + px;
        v8bf v = {};
        if (iy >= 0 && iy < INRES && ix >= 0 && ix < INRES)
          v = *(const v8bf*)&act[(((long)b * INRES + iy) * INRES + ix) * CIN + cg * 8];
        *(v8bf*)&patch[(py * SPAN + px) * CIN + cg * 8] = v;
      }
    }
    __syncthreads();
  }

  const int hi = (lane >> 4) & 1;
  const int ln = lane & 15;
  const v16bf zv = {};

  for (int mp = 0; mp < MPW; ++mp) {
    const int mtile = wave + mp * WAVES;
    const int co0 = mtile << 4;
    for (int pass = 0; pass < NPASS; ++pass) {
      v8f acc[NB];
      #pragma unroll
      for (int j = 0; j < NB; ++j) { v8f z = {}; acc[j] = z; }

      for (int kh = 0; kh < 3; ++kh)
        for (int kw = 0; kw < 3; ++kw) {
          #pragma unroll
          for (int kc = 0; kc < KCH; ++kc) {
            const long tile = ((long)(kh * 3 + kw) * KCH + kc) * MT + mtile;
            v16bf av = *(const v16bf*)&wpack[tile * 512 + lane * 16];
            #pragma unroll
            for (int j = 0; j < NB; ++j) {
              int flat = (pass * NB + j) * 16 + ln;
              int oyl = flat >> TPL, oxl = flat & (TP - 1);
              v16bf bvv;
              if constexpr (USE_LDS) {
                int py = oyl * STRIDE + kh, px = oxl * STRIDE + kw;
                bvv = *(const v16bf*)&patch[(py * SPAN + px) * CIN + kc * 32 + hi * 16];
              } else {
                int iy = (tr * TP + oyl) * STRIDE + kh - 1;
                int ix = (tc * TP + oxl) * STRIDE + kw - 1;
                bvv = zv;
                if (iy >= 0 && iy < INRES && ix >= 0 && ix < INRES)
                  bvv = *(const v16bf*)&act[(((long)b * INRES + iy) * INRES + ix) * CIN
                                            + kc * 32 + hi * 16];
              }
              acc[j] = WMMA_BF16(av, bvv, acc[j]);
            }
          }
        }

      v8f bia = *(const v8f*)&bias[co0 + hi * 8];
      #pragma unroll
      for (int j = 0; j < NB; ++j) {
        int flat = (pass * NB + j) * 16 + ln;
        int oy = tr * TP + (flat >> TPL), ox = tc * TP + (flat & (TP - 1));
        v8bf ov;
        #pragma unroll
        for (int r = 0; r < 8; ++r) {
          float f = acc[j][r] + bia[r];
          ov[r] = (bf16)(f > 0.f ? f : 0.f);
        }
        *(v8bf*)&out[(((long)b * OUTRES + oy) * OUTRES + ox) * COUT + co0 + hi * 8] = ov;
      }
    }
  }
}

// ---------------------------------------------------------------------------
// Up-branch: ConvTranspose k==s (s in {1,2,4}) + bias + ReLU + mask.
// Stride==kernel: each output pixel uses one tap (y%s, x%s) -> pure GEMMs per
// parity class. Block = 8 waves (all 128 couts); input tile async-staged in LDS.
// Writes f32 NCHW slice of d_out (non-temporal: streamed once).
// ---------------------------------------------------------------------------
template<int CIN, int INRES, int SLOG>
__global__ __launch_bounds__(256)
void k_deconv(const bf16* __restrict__ act, const bf16* __restrict__ wpack,
              const float* __restrict__ bias, const int* __restrict__ mask,
              float* __restrict__ dout, int choff) {
  constexpr int S   = 1 << SLOG;
  constexpr int Q   = 16 >> SLOG;          // input tile side = INRES/16
  constexpr int QL  = 4 - SLOG;
  constexpr int SPC = 16 >> (2 * SLOG);    // subtiles per parity class
  constexpr int NB  = (SPC < 4) ? SPC : 4;
  constexpr int PPC = SPC / NB;
  constexpr int KCH = CIN / 32;
  constexpr int MT  = 8;

  __shared__ alignas(32) bf16 patch[Q * Q * CIN];

  const int lane = threadIdx.x;
  const int wave = threadIdx.y;             // == mtile
  const int tid  = wave * 32 + lane;
  const int tileId = blockIdx.x;
  const int b  = tileId >> 8;
  const int tr = (tileId >> 4) & 15;
  const int tc = tileId & 15;

  if (!mask[tileId]) {                      // zero 128 couts x 16x16 px (f32)
    v4f z = {};
    for (int i = tid; i < 128 * 16 * 4; i += 256) {
      int co = i >> 6, row = (i >> 2) & 15, qx = i & 3;
      long a = (((long)b * 384 + choff + co) * 256 + tr * 16 + row) * 256
               + tc * 16 + qx * 4;
      __builtin_nontemporal_store(z, (v4f*)&dout[a]);
    }
    return;
  }

  {                                         // stage input tile (no halo, no OOB)
    constexpr int NCHK = Q * Q * CIN / 8;
    for (int c = tid; c < NCHK; c += 256) {
      int cg  = c % (CIN / 8);
      int pix = c / (CIN / 8);
      int yy = pix >> QL, xx = pix & (Q - 1);
      int iy = tr * Q + yy, ix = tc * Q + xx;
#if ASYNC_LDS
      async_b128(&act[(((long)b * INRES + iy) * INRES + ix) * CIN + cg * 8],
                 &patch[pix * CIN + cg * 8]);
#else
      *(v8bf*)&patch[pix * CIN + cg * 8] =
          *(const v8bf*)&act[(((long)b * INRES + iy) * INRES + ix) * CIN + cg * 8];
#endif
    }
#if ASYNC_LDS
    __builtin_amdgcn_s_wait_asynccnt(0);
#endif
    __syncthreads();
  }

  const int hi = (lane >> 4) & 1;
  const int ln = lane & 15;
  const int mtile = wave;
  const int co0 = mtile << 4;
  const v8f bia = *(const v8f*)&bias[co0 + hi * 8];

  for (int cls = 0; cls < S * S; ++cls) {
    const int py = cls >> SLOG, px = cls & (S - 1);
    for (int pp = 0; pp < PPC; ++pp) {
      v8f acc[NB];
      #pragma unroll
      for (int j = 0; j < NB; ++j) { v8f z = {}; acc[j] = z; }

      #pragma unroll
      for (int kc = 0; kc < KCH; ++kc) {
        v16bf av = *(const v16bf*)&wpack[((long)(cls * KCH + kc) * MT + mtile) * 512
                                         + lane * 16];
        #pragma unroll
        for (int j = 0; j < NB; ++j) {
          int j2 = (pp * NB + j) * 16 + ln;        // index in Q*Q input tile
          int yy = j2 >> QL, xx = j2 & (Q - 1);
          v16bf bvv = *(const v16bf*)&patch[(yy * Q + xx) * CIN + kc * 32 + hi * 16];
          acc[j] = WMMA_BF16(av, bvv, acc[j]);
        }
      }

      #pragma unroll
      for (int j = 0; j < NB; ++j) {
        int j2 = (pp * NB + j) * 16 + ln;
        int yy = j2 >> QL, xx = j2 & (Q - 1);
        int oy = tr * 16 + yy * S + py, ox = tc * 16 + xx * S + px;
        #pragma unroll
        for (int r = 0; r < 8; ++r) {
          float f = acc[j][r] + bia[r];
          f = f > 0.f ? f : 0.f;
          int co = co0 + hi * 8 + r;
          __builtin_nontemporal_store(
              f, &dout[(((long)b * 384 + choff + co) * 256 + oy) * 256 + ox]);
        }
      }
    }
  }
}

// ---------------------------------------------------------------------------
// Host driver
// ---------------------------------------------------------------------------
extern "C" void kernel_launch(void* const* d_in, const int* in_sizes, int n_in,
                              void* d_out, int out_size, void* d_ws, size_t ws_size,
                              hipStream_t stream) {
  (void)n_in; (void)out_size; (void)ws_size;

  static const int cCin[16]  = {64,64,64,64,  64,128,128,128,128,128, 128,256,256,256,256,256};
  static const int cCout[16] = {64,64,64,64, 128,128,128,128,128,128, 256,256,256,256,256,256};
  static const int dCin[3]   = {64,128,256};
  static const int dKs[3]    = {1,2,4};

  long woff[19], boff[19], wacc = 0, bacc = 0;
  for (int i = 0; i < 16; ++i) {
    woff[i] = wacc; boff[i] = bacc;
    wacc += (long)cCout[i] * cCin[i] * 9; bacc += cCout[i];
  }
  for (int i = 0; i < 3; ++i) {
    woff[16 + i] = wacc; boff[16 + i] = bacc;
    wacc += 128L * dCin[i] * dKs[i] * dKs[i]; bacc += 128;
  }

  char* ws = (char*)d_ws;
  int*   mask  = (int*)(ws + 0);
  float* bias  = (float*)(ws + 4096);
  bf16*  wpack = (bf16*)(ws + 32768);
  bf16*  A = (bf16*)(ws + 32768 + (16L << 20));               // 67.1 MB buffer
  bf16*  B = (bf16*)(ws + 32768 + (16L << 20) + (72L << 20)); // 16.8 MB buffer

  const int*   coords = (const int*)d_in[96];
  const int    ncoord = in_sizes[96];
  const float* x_in   = (const float*)d_in[0];

  k_build_mask<<<dim3(16), dim3(32), 0, stream>>>(coords, ncoord, mask);
  k_convert_in<<<dim3(16384), dim3(256), 0, stream>>>(x_in, A);

  for (int i = 0; i < 16; ++i) {
    int pb = 1 + 5 * i;   // leaves per layer dict: b,g,m,v,w
    int total = cCout[i] * cCin[i] * 9;
    k_pack<<<dim3((total + 255) / 256), dim3(256), 0, stream>>>(
        (const float*)d_in[pb + 4], (const float*)d_in[pb + 1],
        (const float*)d_in[pb + 0], (const float*)d_in[pb + 2],
        (const float*)d_in[pb + 3],
        wpack + woff[i], bias + boff[i], cCout[i], cCin[i], 3);
  }
  for (int i = 0; i < 3; ++i) {
    int pb = 81 + 5 * i;
    int total = 128 * dCin[i] * dKs[i] * dKs[i];
    k_pack<<<dim3((total + 255) / 256), dim3(256), 0, stream>>>(
        (const float*)d_in[pb + 4], (const float*)d_in[pb + 1],
        (const float*)d_in[pb + 0], (const float*)d_in[pb + 2],
        (const float*)d_in[pb + 3],
        wpack + woff[16 + i], bias + boff[16 + i], 128, dCin[i], dKs[i]);
  }

  float* dout = (float*)d_out;
  const dim3 G(512);

  // ---- block 0 (res 256, C 64) ----
  k_conv< 64, 64,512,256,2,false><<<G, dim3(32,4), 0, stream>>>(A, wpack+woff[0],  bias+boff[0],  mask, B);
  k_conv< 64, 64,256,256,1,true ><<<G, dim3(32,4), 0, stream>>>(B, wpack+woff[1],  bias+boff[1],  mask, A);
  k_conv< 64, 64,256,256,1,true ><<<G, dim3(32,4), 0, stream>>>(A, wpack+woff[2],  bias+boff[2],  mask, B);
  k_conv< 64, 64,256,256,1,true ><<<G, dim3(32,4), 0, stream>>>(B, wpack+woff[3],  bias+boff[3],  mask, A);
  k_deconv< 64,256,0><<<G, dim3(32,8), 0, stream>>>(A, wpack+woff[16], bias+boff[16], mask, dout, 0);

  // ---- block 1 (res 128, C 128) ----
  k_conv< 64,128,256,128,2,true ><<<G, dim3(32,8), 0, stream>>>(A, wpack+woff[4],  bias+boff[4],  mask, B);
  k_conv<128,128,128,128,1,true ><<<G, dim3(32,8), 0, stream>>>(B, wpack+woff[5],  bias+boff[5],  mask, A);
  k_conv<128,128,128,128,1,true ><<<G, dim3(32,8), 0, stream>>>(A, wpack+woff[6],  bias+boff[6],  mask, B);
  k_conv<128,128,128,128,1,true ><<<G, dim3(32,8), 0, stream>>>(B, wpack+woff[7],  bias+boff[7],  mask, A);
  k_conv<128,128,128,128,1,true ><<<G, dim3(32,8), 0, stream>>>(A, wpack+woff[8],  bias+boff[8],  mask, B);
  k_conv<128,128,128,128,1,true ><<<G, dim3(32,8), 0, stream>>>(B, wpack+woff[9],  bias+boff[9],  mask, A);
  k_deconv<128,128,1><<<G, dim3(32,8), 0, stream>>>(A, wpack+woff[17], bias+boff[17], mask, dout, 128);

  // ---- block 2 (res 64, C 256) ----
  k_conv<128,256,128, 64,2,true ><<<G, dim3(32,8), 0, stream>>>(A, wpack+woff[10], bias+boff[10], mask, B);
  k_conv<256,256, 64, 64,1,true ><<<G, dim3(32,8), 0, stream>>>(B, wpack+woff[11], bias+boff[11], mask, A);
  k_conv<256,256, 64, 64,1,true ><<<G, dim3(32,8), 0, stream>>>(A, wpack+woff[12], bias+boff[12], mask, B);
  k_conv<256,256, 64, 64,1,true ><<<G, dim3(32,8), 0, stream>>>(B, wpack+woff[13], bias+boff[13], mask, A);
  k_conv<256,256, 64, 64,1,true ><<<G, dim3(32,8), 0, stream>>>(A, wpack+woff[14], bias+boff[14], mask, B);
  k_conv<256,256, 64, 64,1,true ><<<G, dim3(32,8), 0, stream>>>(B, wpack+woff[15], bias+boff[15], mask, A);
  k_deconv<256, 64,2><<<G, dim3(32,8), 0, stream>>>(A, wpack+woff[18], bias+boff[18], mask, dout, 256);
}
